// GCNWithCoarsening_86277303042080
// MI455X (gfx1250) — compile-verified, matrix operand order checked by reading, not run on hardware
//
#include <hip/hip_runtime.h>
#include <hip/hip_bf16.h>
#include <math.h>

typedef __attribute__((ext_vector_type(16))) _Float16 v16h;
typedef __attribute__((ext_vector_type(8)))  float    v8f;

#define N_NODES  50000
#define N_EDGES  800000
#define CH       128
#define N_GRAPHS 64
#define NF       (N_NODES * CH)          // 6,400,000 floats per feature map
#define BN_EPS   1e-5f
#define MTILES_PER_BLOCK 5               // 3125 tiles = 625 blocks * 5

__device__ __forceinline__ float gelu_exact(float x) {
    return 0.5f * x * (1.0f + erff(x * 0.70710678118654752440f));
}

// ---------------- utility kernels ----------------
__global__ void fill_f32(float* __restrict__ p, float v, int n) {
    int i = blockIdx.x * blockDim.x + threadIdx.x;
    if (i < n) p[i] = v;
}

__global__ void f32_to_f16(const float* __restrict__ in, _Float16* __restrict__ out, int n) {
    int i = blockIdx.x * blockDim.x + threadIdx.x;
    if (i < n) out[i] = (_Float16)in[i];
}

__global__ void deg_accum(const int* __restrict__ dst, float* __restrict__ deg, int nE) {
    int e = blockIdx.x * blockDim.x + threadIdx.x;
    if (e < nE) atomicAdd(&deg[dst[e]], 1.0f);
}

__global__ void dinv_kernel(float* __restrict__ deg, int n) {
    int i = blockIdx.x * blockDim.x + threadIdx.x;
    if (i < n) {
        float d = deg[i];
        deg[i] = (d > 0.0f) ? rsqrtf(d) : 0.0f;
    }
}

// ---------------- weight pre-pack: W[128x128] f32 -> B fragments f16 ----------------
// Output layout: [ntile(8)][kstep(4)][lane(32)][8 dwords]  == 32 KB total.
// Fragment layout matches 16-bit B operand: lane holds col n = ntile*16 + (lane&15),
// VGPR j<4 -> K = kb+2j (+lo/hi pair), j>=4 -> K = kb+16+2(j-4), kb = (lane>=16)*8.
__global__ void pack_w(const float* __restrict__ W, unsigned* __restrict__ out) {
    int t = blockIdx.x * blockDim.x + threadIdx.x;   // 0..1023
    if (t >= 8 * 4 * 32) return;
    int lane = t & 31;
    int ks   = (t >> 5) & 3;
    int nt   = t >> 7;
    int r  = lane & 15;
    int kb = (lane >> 4) << 3;
    int n  = nt * 16 + r;
    #pragma unroll
    for (int j = 0; j < 8; ++j) {
        int k = ks * 32 + kb + ((j < 4) ? (2 * j) : (16 + 2 * (j - 4)));
        union { _Float16 h[2]; unsigned u; } p;
        p.h[0] = (_Float16)W[(size_t)k       * CH + n];
        p.h[1] = (_Float16)W[(size_t)(k + 1) * CH + n];
        out[(size_t)t * 8 + j] = p.u;
    }
}

// ---------------- WMMA GEMM: C[M x 128] = A[M x 128] @ W[128 x 128] ----------------
// 8 waves/block; wave w owns cols [16w,16w+16). Each wave loads its 4 B fragments
// into registers once (2 x b128 each), then runs 5 M-tiles x 4 WMMAs reusing them.
// A fragments: two contiguous 16-byte loads per K-step (layout-exact).
__global__ void wmma_gemm_128(const _Float16* __restrict__ A,
                              const uint4* __restrict__ Wpack,
                              float* __restrict__ C) {
    const int wave = threadIdx.x >> 5;
    const int lane = threadIdx.x & 31;
    const int r  = lane & 15;
    const int kb = (lane >> 4) << 3;     // 0 or 8
    const int n0 = wave * 16;

    union frag { v16h v; uint4 q[2]; };

    // load B fragments for all 4 K-steps (register-resident)
    frag b[4];
    #pragma unroll
    for (int ks = 0; ks < 4; ++ks) {
        const uint4* wp = Wpack + (size_t)(((wave * 4 + ks) * 32 + lane) * 2);
        b[ks].q[0] = wp[0];
        b[ks].q[1] = wp[1];
    }

    const int mb = (lane >> 4) << 3;
    const int nc = n0 + r;

    for (int mt = 0; mt < MTILES_PER_BLOCK; ++mt) {
        const int m0 = (blockIdx.x * MTILES_PER_BLOCK + mt) * 16;
        const _Float16* arow = A + (size_t)(m0 + r) * CH;
        v8f acc = {};
        #pragma unroll
        for (int ks = 0; ks < 4; ++ks) {
            frag a;
            a.q[0] = *reinterpret_cast<const uint4*>(arow + ks * 32 + kb);       // K = kb..kb+7
            a.q[1] = *reinterpret_cast<const uint4*>(arow + ks * 32 + kb + 16);  // K = kb+16..kb+23
            acc = __builtin_amdgcn_wmma_f32_16x16x32_f16(
                      false, a.v, false, b[ks].v, (short)0, acc, false, false);
        }
        // C/D layout: VGPR v -> M = v + (lane>=16 ? 8:0), N = lane&15
        #pragma unroll
        for (int v = 0; v < 8; ++v)
            C[(size_t)(m0 + mb + v) * CH + nc] = acc[v];
    }
}

// ---------------- edge scatter: agg[dst] += hw[src] * dinv[src]*dinv[dst] ----------------
__global__ void edge_scatter(const float* __restrict__ hw,
                             const int* __restrict__ src, const int* __restrict__ dst,
                             const float* __restrict__ dinv, float* __restrict__ agg) {
    int tid = blockIdx.x * blockDim.x + threadIdx.x;
    const int total = (N_EDGES + N_NODES) * (CH / 4);
    if (tid >= total) return;
    int e  = tid >> 5;             // 32 threads per edge
    int c0 = (tid & 31) << 2;      // channel group of 4
    int s, d;
    if (e < N_EDGES) { s = src[e]; d = dst[e]; }
    else             { s = d = e - N_EDGES; }   // self-loop
    float w = dinv[s] * dinv[d];
    const float4 v = *reinterpret_cast<const float4*>(hw + (size_t)s * CH + c0);
    float* base = agg + (size_t)d * CH + c0;
    atomicAdd(base + 0, v.x * w);
    atomicAdd(base + 1, v.y * w);
    atomicAdd(base + 2, v.z * w);
    atomicAdd(base + 3, v.w * w);
}

__global__ void bias_add(float* __restrict__ h, const float* __restrict__ b, int total) {
    int i = blockIdx.x * blockDim.x + threadIdx.x;
    if (i < total) h[i] += b[i & (CH - 1)];
}

// ---------------- BatchNorm stats + fused BN+GELU ----------------
__global__ void stats_kernel(const float* __restrict__ h, float* __restrict__ stats) {
    int c = threadIdx.x;                     // one thread per channel
    int start = blockIdx.x * 256;
    int end = start + 256; if (end > N_NODES) end = N_NODES;
    float s = 0.0f, s2 = 0.0f;
    for (int i = start; i < end; ++i) {
        float v = h[(size_t)i * CH + c];
        s += v; s2 += v * v;
    }
    atomicAdd(&stats[c],      s);
    atomicAdd(&stats[CH + c], s2);
}

__global__ void bn_gelu_kernel(float* __restrict__ h, const float* __restrict__ stats,
                               const float* __restrict__ g, const float* __restrict__ beta) {
    int i = blockIdx.x * blockDim.x + threadIdx.x;
    if (i >= NF) return;
    int c = i & (CH - 1);
    const float inv_n = 1.0f / (float)N_NODES;
    float mean = stats[c] * inv_n;
    float var  = stats[CH + c] * inv_n - mean * mean;
    float x = (h[i] - mean) * rsqrtf(var + BN_EPS);
    x = g[c] * x + beta[c];
    h[i] = gelu_exact(x);
}

// ---------------- global mean pool ----------------
__global__ void pool_accum(const float* __restrict__ h, const int* __restrict__ batch,
                           float* __restrict__ pooled, float* __restrict__ cnt) {
    int tid = blockIdx.x * blockDim.x + threadIdx.x;
    if (tid >= NF) return;
    int n = tid >> 7;
    int c = tid & (CH - 1);
    int g = batch[n];
    atomicAdd(&pooled[(size_t)g * CH + c], h[tid]);
    if (c == 0) atomicAdd(&cnt[g], 1.0f);
}

__global__ void pool_div(float* __restrict__ pooled, const float* __restrict__ cnt) {
    int i = blockIdx.x * blockDim.x + threadIdx.x;
    if (i >= N_GRAPHS * CH) return;
    pooled[i] /= fmaxf(cnt[i >> 7], 1.0f);
}

// ---------------- small MLP head ----------------
__global__ void head_linear(const float* __restrict__ in, const float* __restrict__ W,
                            const float* __restrict__ b, float* __restrict__ out,
                            int Fout, int do_gelu) {
    int g = blockIdx.x;
    int j = threadIdx.x;
    if (j >= Fout) return;
    float s = b[j];
    for (int k = 0; k < CH; ++k)
        s += in[(size_t)g * CH + k] * W[(size_t)k * Fout + j];
    if (do_gelu) s = gelu_exact(s);
    out[(size_t)g * Fout + j] = s;
}

// ---------------- launch ----------------
extern "C" void kernel_launch(void* const* d_in, const int* in_sizes, int n_in,
                              void* d_out, int out_size, void* d_ws, size_t ws_size,
                              hipStream_t stream) {
    (void)in_sizes; (void)n_in; (void)out_size; (void)ws_size;

    const float* x    = (const float*)d_in[0];
    const int*   ei   = (const int*)d_in[1];       // [2, N_EDGES]
    const int*   batc = (const int*)d_in[2];
    // d_in[3] = num_graphs (constant 64)
    const float* Wb0 = (const float*)d_in[4];  const float* bb0 = (const float*)d_in[5];
    const float* gb0 = (const float*)d_in[6];  const float* beb0= (const float*)d_in[7];
    const float* Wb1 = (const float*)d_in[8];  const float* bb1 = (const float*)d_in[9];
    const float* Wa0 = (const float*)d_in[10]; const float* ba0 = (const float*)d_in[11];
    const float* ga0 = (const float*)d_in[12]; const float* bea0= (const float*)d_in[13];
    const float* Wa1 = (const float*)d_in[14]; const float* ba1 = (const float*)d_in[15];
    const float* ga1 = (const float*)d_in[16]; const float* bea1= (const float*)d_in[17];
    const float* Wa2 = (const float*)d_in[18]; const float* ba2 = (const float*)d_in[19];
    const float* Wh0 = (const float*)d_in[20]; const float* bh0 = (const float*)d_in[21];
    const float* Wh1 = (const float*)d_in[22]; const float* bh1 = (const float*)d_in[23];
    const float* Wh2 = (const float*)d_in[24]; const float* bh2 = (const float*)d_in[25];

    const int* srcp = ei;
    const int* dstp = ei + N_EDGES;

    // workspace carve-up (256B aligned)
    char* ws = (char*)d_ws;
    size_t off = 0;
    auto carve = [&](size_t bytes) -> void* {
        void* p = ws + off;
        off += (bytes + 255) & ~(size_t)255;
        return p;
    };
    float*    dinv   = (float*)   carve((size_t)N_NODES * 4);
    float*    hA     = (float*)   carve((size_t)NF * 4);
    float*    hB     = (float*)   carve((size_t)NF * 4);
    float*    hw     = (float*)   carve((size_t)NF * 4);
    _Float16* hf16   = (_Float16*)carve((size_t)NF * 2);
    unsigned* wpack  = (unsigned*)carve((size_t)8 * 4 * 32 * 8 * 4);   // 32 KB
    float*    stats  = (float*)   carve(2 * CH * 4);
    float*    pooled = (float*)   carve((size_t)N_GRAPHS * CH * 4);
    float*    cnt    = (float*)   carve((size_t)N_GRAPHS * 4);
    float*    t0     = (float*)   carve((size_t)N_GRAPHS * CH * 4);
    float*    t1     = (float*)   carve((size_t)N_GRAPHS * CH * 4);

    const int TB = 256;
    auto blocks = [](long n, int tb) { return (int)((n + tb - 1) / tb); };

    // 1) degrees (self-loop => init 1.0) -> dinv
    fill_f32<<<blocks(N_NODES, TB), TB, 0, stream>>>(dinv, 1.0f, N_NODES);
    deg_accum<<<blocks(N_EDGES, TB), TB, 0, stream>>>(dstp, dinv, N_EDGES);
    dinv_kernel<<<blocks(N_NODES, TB), TB, 0, stream>>>(dinv, N_NODES);

    const int scatter_blocks = blocks((long)(N_EDGES + N_NODES) * (CH / 4), TB);
    const int gemm_blocks = (N_NODES / 16) / MTILES_PER_BLOCK;   // 625

    auto conv = [&](const float* hin, const float* Wd, const float* bd, float* outBuf) {
        f32_to_f16<<<blocks(NF, TB), TB, 0, stream>>>(hin, hf16, NF);
        pack_w<<<4, 256, 0, stream>>>(Wd, wpack);
        wmma_gemm_128<<<gemm_blocks, TB, 0, stream>>>(hf16, (const uint4*)wpack, hw);
        fill_f32<<<blocks(NF, TB), TB, 0, stream>>>(outBuf, 0.0f, NF);
        edge_scatter<<<scatter_blocks, TB, 0, stream>>>(hw, srcp, dstp, dinv, outBuf);
        bias_add<<<blocks(NF, TB), TB, 0, stream>>>(outBuf, bd, NF);
    };
    auto bn = [&](float* hbuf, const float* g, const float* beta) {
        fill_f32<<<1, 256, 0, stream>>>(stats, 0.0f, 2 * CH);
        stats_kernel<<<blocks(N_NODES, 256), CH, 0, stream>>>(hbuf, stats);
        bn_gelu_kernel<<<blocks(NF, TB), TB, 0, stream>>>(hbuf, stats, g, beta);
    };

    // 2) five GCN layers
    conv(x,  Wb0, bb0, hA); bn(hA, gb0, beb0);          // conv -> bn -> gelu
    conv(hA, Wb1, bb1, hB);                             // conv
    conv(hB, Wa0, ba0, hA); bn(hA, ga0, bea0);          // conv -> bn -> gelu
    conv(hA, Wa1, ba1, hB); bn(hB, ga1, bea1);          // conv -> bn -> gelu
    conv(hB, Wa2, ba2, hA);                             // conv

    // 3) global mean pool
    fill_f32<<<blocks(N_GRAPHS * CH, TB), TB, 0, stream>>>(pooled, 0.0f, N_GRAPHS * CH);
    fill_f32<<<1, N_GRAPHS, 0, stream>>>(cnt, 0.0f, N_GRAPHS);
    pool_accum<<<blocks(NF, TB), TB, 0, stream>>>(hA, batc, pooled, cnt);
    pool_div<<<blocks(N_GRAPHS * CH, TB), TB, 0, stream>>>(pooled, cnt);

    // 4) MLP head
    head_linear<<<N_GRAPHS, CH, 0, stream>>>(pooled, Wh0, bh0, t0, CH, 1);
    head_linear<<<N_GRAPHS, CH, 0, stream>>>(t0,     Wh1, bh1, t1, CH, 1);
    head_linear<<<N_GRAPHS, 16, 0, stream>>>(t1,     Wh2, bh2, (float*)d_out, 10, 0);
}
